// DisplacementNet_25031069401172
// MI455X (gfx1250) — matrix operand order repaired and checked by reading, not compiled, not run on hardware
//
#include <hip/hip_runtime.h>
#include <math.h>

// ---------------------------------------------------------------------------
// DisplacementNet on MI455X (gfx1250): WMMA f16->f32 GEMM pipeline + VALU kNN
// B-operand staged into LDS via CDNA5 async copy (global_load_async_to_lds).
// ---------------------------------------------------------------------------

typedef _Float16 half_t;
typedef __attribute__((ext_vector_type(16))) _Float16 v16h;
typedef __attribute__((ext_vector_type(8)))  float    v8f;

#define NODES   65536
#define KNN_K   12
#define WIDTH   256
#define GWIDTH  192
#define DEPTH   7
#define GLAYERS 4
#define CONDD   64
#define FFPAD   64   // 48 fourier features padded to K-multiple of 32

union F16Frag { v16h v; uint4 q[2]; };

// silu with hardware v_rcp_f32 (result feeds an f16 store; ~1ulp f32 is ample)
__device__ __forceinline__ float silu_f(float x) {
    return x * __builtin_amdgcn_rcpf(1.0f + __expf(-x));
}

// Raw LDS byte offset of a generic pointer to __shared__ (LDS aperture keeps
// the offset in addr[31:0]).
__device__ __forceinline__ unsigned lds_offset(const void* p) {
    return (unsigned)(uintptr_t)p;
}

// ---------------------------------------------------------------------------
// kNN: one row per thread, 256-point LDS tiles, register-resident 12-best list
// ---------------------------------------------------------------------------
__global__ void knn_kernel(const float* __restrict__ x, int* __restrict__ knn, int n) {
    __shared__ float sx[256], sy[256], sz[256];
    const int tid = threadIdx.x;
    const int i = blockIdx.x * 256 + tid;
    const float xi = x[3 * i], yi = x[3 * i + 1], zi = x[3 * i + 2];

    float bd[KNN_K];
    int   bi[KNN_K];
#pragma unroll
    for (int s = 0; s < KNN_K; ++s) { bd[s] = 3.0e38f; bi[s] = 0; }

    for (int j0 = 0; j0 < n; j0 += 256) {
        const int j = j0 + tid;
        sx[tid] = x[3 * j];
        sy[tid] = x[3 * j + 1];
        sz[tid] = x[3 * j + 2];
        __syncthreads();
        if (j0 + 256 < n) __builtin_prefetch(&x[3 * (j0 + 256 + tid)], 0, 1);  // global_prefetch_b8
#pragma unroll 4
        for (int jj = 0; jj < 256; ++jj) {
            const float dx = xi - sx[jj], dy = yi - sy[jj], dz = zi - sz[jj];
            float d = dx * dx + dy * dy + dz * dz;
            const int cj = j0 + jj;
            if (cj == i) d = 3.0e38f;          // exclude self
            if (d < bd[KNN_K - 1]) {
                bd[KNN_K - 1] = d; bi[KNN_K - 1] = cj;
#pragma unroll
                for (int s = KNN_K - 1; s > 0; --s) {
                    if (bd[s] < bd[s - 1]) {
                        float td = bd[s]; bd[s] = bd[s - 1]; bd[s - 1] = td;
                        int   ti = bi[s]; bi[s] = bi[s - 1]; bi[s - 1] = ti;
                    }
                }
            }
        }
        __syncthreads();
    }
#pragma unroll
    for (int s = 0; s < KNN_K; ++s) knn[i * KNN_K + s] = bi[s];
}

// ---------------------------------------------------------------------------
// Fourier features -> f16, K padded 48 -> 64 with zeros
// ---------------------------------------------------------------------------
__global__ void fourier_kernel(const float* __restrict__ x, const float* __restrict__ Bm,
                               half_t* __restrict__ ffh, int n) {
    const int i = blockIdx.x * 256 + threadIdx.x;
    if (i >= n) return;
    const float x0 = x[3 * i], x1 = x[3 * i + 1], x2 = x[3 * i + 2];
    half_t* row = ffh + (size_t)i * FFPAD;
#pragma unroll
    for (int s = 0; s < 3; ++s) {
#pragma unroll
        for (int m = 0; m < 8; ++m) {
            const float proj = x0 * Bm[s * 24 + 0 * 8 + m] +
                               x1 * Bm[s * 24 + 1 * 8 + m] +
                               x2 * Bm[s * 24 + 2 * 8 + m];
            float sv, cv;
            __sincosf(proj, &sv, &cv);
            row[s * 8 + m]      = (half_t)sv;
            row[24 + s * 8 + m] = (half_t)cv;
        }
    }
#pragma unroll
    for (int c = 48; c < FFPAD; ++c) row[c] = (half_t)0.0f;
}

// ---------------------------------------------------------------------------
// Weight prepack into WMMA B-fragment layout:
//   dst[((kt*NT + nt)*32 + lane)*16 + t], lane -> col nt*16+(lane&15),
//   K = kt*32 + (lane>>4)*16 + t.  Supports stacked [W1;W2] and zero K-pad.
// ---------------------------------------------------------------------------
__global__ void pack_weights(const float* __restrict__ W1, int k1,
                             const float* __restrict__ W2, int k2,
                             int ncols, int kpad, half_t* __restrict__ dst) {
    const int idx = blockIdx.x * 256 + threadIdx.x;
    const int total = (kpad / 32) * (ncols / 16) * 512;
    if (idx >= total) return;
    const int t    = idx & 15;
    const int lane = (idx >> 4) & 31;
    const int tile = idx >> 9;
    const int NT = ncols / 16;
    const int nt = tile % NT, kt = tile / NT;
    const int col = nt * 16 + (lane & 15);
    const int kk  = kt * 32 + (lane >> 4) * 16 + t;
    float v = 0.0f;
    if (kk < k1)            v = W1[(size_t)kk * ncols + col];
    else if (kk < k1 + k2)  v = W2[(size_t)(kk - k1) * ncols + col];
    dst[idx] = (half_t)v;
}

// ---------------------------------------------------------------------------
// FiLM params: gamma/beta[l] = cond @ Wf + bf   (7 layers x 256)
// ---------------------------------------------------------------------------
__global__ void film_kernel(const float* __restrict__ cond,
                            const float* __restrict__ Wf_g, const float* __restrict__ bf_g,
                            const float* __restrict__ Wf_b, const float* __restrict__ bf_b,
                            float* __restrict__ gb) {
    const int l = blockIdx.x, t = threadIdx.x;
    float g = bf_g[l * WIDTH + t];
    float b = bf_b[l * WIDTH + t];
    for (int c = 0; c < CONDD; ++c) {
        const float cv = cond[c];
        g += cv * Wf_g[((size_t)l * CONDD + c) * WIDTH + t];
        b += cv * Wf_b[((size_t)l * CONDD + c) * WIDTH + t];
    }
    gb[l * 512 + t]       = g;
    gb[l * 512 + 256 + t] = b;
}

// ---------------------------------------------------------------------------
// Neighbor mean: agg (cols 192..383) = mean over 12 neighbors of feat (cols 0..191)
// ---------------------------------------------------------------------------
__global__ void gather_mean(const int* __restrict__ knn, half_t* __restrict__ g) {
    const int i = blockIdx.x;
    const int c = threadIdx.x;              // 192 threads
    const int* ki = knn + (size_t)i * KNN_K;
    float acc = 0.0f;
#pragma unroll
    for (int j = 0; j < KNN_K; ++j)
        acc += (float)g[(size_t)ki[j] * 384 + c];
    g[(size_t)i * 384 + GWIDTH + c] = (half_t)(acc * (1.0f / 12.0f));
}

// ---------------------------------------------------------------------------
// WMMA GEMM: Out(f16) = epilogue(A(f16, row-major) @ Bpacked + bias)
//   wave computes 16 rows x 64 cols (4 fragments); block = 8 waves = 128 rows.
//   The block's B strip (KT k-tiles x 64 cols) is staged into LDS once with
//   CDNA5 async copies, then fragments come from ds_load_b128.
//   ACT: silu, FILM: *(1+gamma)+beta before act, POSTADD: += addend after act.
// ---------------------------------------------------------------------------
template <int KT, int ACT, bool FILM, bool POSTADD>
__global__ void gemm_wmma(const half_t* __restrict__ A, int lda,
                          const half_t* __restrict__ Bp,
                          const float* __restrict__ bias,
                          const float* __restrict__ gamma,
                          const float* __restrict__ beta,
                          const half_t* __restrict__ addend, int ldadd,
                          half_t* __restrict__ Out, int ldo,
                          int NT) {
    __shared__ half_t sB[KT * 4 * 512];   // [kt][t][lane*16 .. +15]

    const int lane = threadIdx.x & 31;
    const int wave = threadIdx.x >> 5;
    const int row0 = (blockIdx.y * 8 + wave) * 16;
    const int ntBase = blockIdx.x * 4;

    // --- async-stage the B strip: KT*4 tiles * 1KB; one b128 (16B) per issue.
    {
        const int tid = threadIdx.x;
        for (int q = tid; q < KT * 256; q += 256) {
            const int c2 = q & 1;
            const int ln = (q >> 1) & 31;
            const int t  = (q >> 6) & 3;
            const int kt = q >> 8;
            const half_t* gsrc =
                Bp + ((((size_t)kt * NT + ntBase + t) * 32 + ln) * 16 + c2 * 8);
            const unsigned laddr =
                lds_offset(&sB[(kt * 4 + t) * 512 + ln * 16 + c2 * 8]);
            asm volatile("global_load_async_to_lds_b128 %0, %1, off"
                         :: "v"(laddr), "v"(gsrc) : "memory");
        }
        asm volatile("s_wait_asynccnt 0x0" ::: "memory");
        __syncthreads();
    }

    const int m  = lane & 15;
    const int kh = lane >> 4;
    const half_t* arow = A + (size_t)(row0 + m) * lda + kh * 16;

    v8f acc[4] = {};
    for (int kt = 0; kt < KT; ++kt) {
        F16Frag fa;
        fa.q[0] = *(const uint4*)(arow + kt * 32);
        fa.q[1] = *(const uint4*)(arow + kt * 32 + 8);
#pragma unroll
        for (int t = 0; t < 4; ++t) {
            const half_t* bp = &sB[(kt * 4 + t) * 512 + lane * 16];
            F16Frag fb;
            fb.q[0] = *(const uint4*)(bp);
            fb.q[1] = *(const uint4*)(bp + 8);
            acc[t] = __builtin_amdgcn_wmma_f32_16x16x32_f16(
                false, fa.v, false, fb.v, (short)0, acc[t], false, false);
        }
    }

    const int n  = lane & 15;
    const int mh = lane >> 4;
#pragma unroll
    for (int t = 0; t < 4; ++t) {
        const int c = (ntBase + t) * 16 + n;
        const float bia = bias ? bias[c] : 0.0f;
        float gg = 0.0f, bb = 0.0f;
        if (FILM) { gg = gamma[c]; bb = beta[c]; }
#pragma unroll
        for (int v = 0; v < 8; ++v) {
            const int r = row0 + mh * 8 + v;
            float val = acc[t][v] + bia;
            if (FILM) val = val * (1.0f + gg) + bb;
            if (ACT)  val = silu_f(val);
            if (POSTADD) val += (float)addend[(size_t)r * ldadd + c];
            Out[(size_t)r * ldo + c] = (half_t)val;
        }
    }
}

// ---------------------------------------------------------------------------
// Head: out = (h @ W_out + b_out) * 0.01
// ---------------------------------------------------------------------------
__global__ void out_kernel(const half_t* __restrict__ h, const float* __restrict__ Wout,
                           const float* __restrict__ bout, float* __restrict__ out, int n) {
    const int i = blockIdx.x * 256 + threadIdx.x;
    if (i >= n) return;
    float a0 = bout[0], a1 = bout[1], a2 = bout[2];
    const half_t* hr = h + (size_t)i * WIDTH;
    for (int c = 0; c < WIDTH; ++c) {
        const float v = (float)hr[c];
        a0 += v * Wout[c * 3 + 0];
        a1 += v * Wout[c * 3 + 1];
        a2 += v * Wout[c * 3 + 2];
    }
    out[3 * i + 0] = a0 * 0.01f;
    out[3 * i + 1] = a1 * 0.01f;
    out[3 * i + 2] = a2 * 0.01f;
}

// ---------------------------------------------------------------------------
extern "C" void kernel_launch(void* const* d_in, const int* in_sizes, int n_in,
                              void* d_out, int out_size, void* d_ws, size_t ws_size,
                              hipStream_t stream) {
    const float* x      = (const float*)d_in[0];
    const float* cond   = (const float*)d_in[1];
    const float* Bm     = (const float*)d_in[2];
    const float* W_in   = (const float*)d_in[3];
    const float* b_in   = (const float*)d_in[4];
    const float* Wg_in  = (const float*)d_in[5];
    const float* bg_in  = (const float*)d_in[6];
    const float* Ws     = (const float*)d_in[7];
    const float* Wn     = (const float*)d_in[8];
    const float* bg     = (const float*)d_in[9];
    const float* Wg_out = (const float*)d_in[10];
    const float* W      = (const float*)d_in[11];
    const float* bmlp   = (const float*)d_in[12];
    const float* Wf_g   = (const float*)d_in[13];
    const float* bf_g   = (const float*)d_in[14];
    const float* Wf_b   = (const float*)d_in[15];
    const float* bf_b   = (const float*)d_in[16];
    const float* W_out  = (const float*)d_in[17];
    const float* b_out  = (const float*)d_in[18];
    float* out = (float*)d_out;
    (void)in_sizes; (void)n_in; (void)out_size; (void)ws_size;

    const int N = NODES;
    size_t off = 0;
    auto take = [&](size_t bytes) -> char* {
        char* p = (char*)d_ws + off;
        off += (bytes + 255) & ~(size_t)255;
        return p;
    };
    int*    knn  = (int*)   take((size_t)N * KNN_K * 4);
    half_t* ffh  = (half_t*)take((size_t)N * FFPAD * 2);
    half_t* g0   = (half_t*)take((size_t)N * 384 * 2);
    half_t* g1   = (half_t*)take((size_t)N * 384 * 2);
    half_t* h0h  = (half_t*)take((size_t)N * WIDTH * 2);
    half_t* hB   = (half_t*)take((size_t)N * WIDTH * 2);
    float*  gb   = (float*) take(7 * 512 * 4);
    half_t* pWin  = (half_t*)take((size_t)2 * 16 * 512 * 2);      // K64  x 256
    half_t* pWgin = (half_t*)take((size_t)2 * 12 * 512 * 2);      // K64  x 192
    half_t* pG    = (half_t*)take((size_t)4 * 12 * 12 * 512 * 2); // K384 x 192 per layer
    half_t* pWgo  = (half_t*)take((size_t)6 * 16 * 512 * 2);      // K192 x 256
    half_t* pW    = (half_t*)take((size_t)7 * 8 * 16 * 512 * 2);  // K256 x 256 per layer
    half_t* hA = g1;  // alias: graph ping-pong buffer is dead once inject runs

    const dim3 blk(256);

    // graph structure + features
    knn_kernel    <<<N / 256, blk, 0, stream>>>(x, knn, N);
    fourier_kernel<<<N / 256, blk, 0, stream>>>(x, Bm, ffh, N);

    // weight packing
    auto pack = [&](const float* W1, int k1, const float* W2, int k2,
                    int ncols, int kpad, half_t* dst) {
        const int total = (kpad / 32) * (ncols / 16) * 512;
        pack_weights<<<(total + 255) / 256, blk, 0, stream>>>(W1, k1, W2, k2, ncols, kpad, dst);
    };
    pack(W_in, 48, nullptr, 0, WIDTH, FFPAD, pWin);
    pack(Wg_in, 48, nullptr, 0, GWIDTH, FFPAD, pWgin);
    for (int l = 0; l < GLAYERS; ++l)
        pack(Ws + (size_t)l * GWIDTH * GWIDTH, GWIDTH,
             Wn + (size_t)l * GWIDTH * GWIDTH, GWIDTH,
             GWIDTH, 384, pG + (size_t)l * 12 * 12 * 512);
    pack(Wg_out, GWIDTH, nullptr, 0, WIDTH, GWIDTH, pWgo);
    for (int l = 0; l < DEPTH; ++l)
        pack(W + (size_t)l * WIDTH * WIDTH, WIDTH, nullptr, 0,
             WIDTH, WIDTH, pW + (size_t)l * 8 * 16 * 512);

    film_kernel<<<DEPTH, blk, 0, stream>>>(cond, Wf_g, bf_g, Wf_b, bf_b, gb);

    // embeddings: gfeat0 = silu(ff@Wg_in+bg_in), h0 = silu(ff@W_in+b_in)
    gemm_wmma<2, 1, false, false><<<dim3(3, N / 128), blk, 0, stream>>>(
        ffh, FFPAD, pWgin, bg_in, nullptr, nullptr, nullptr, 0, g0, 384, 12);
    gemm_wmma<2, 1, false, false><<<dim3(4, N / 128), blk, 0, stream>>>(
        ffh, FFPAD, pWin, b_in, nullptr, nullptr, nullptr, 0, h0h, WIDTH, 16);
    gather_mean<<<N, GWIDTH, 0, stream>>>(knn, g0);

    // graph layers: fused [gfeat|agg] @ [Ws;Wn]  (K=384)
    for (int l = 0; l < GLAYERS; ++l) {
        half_t* gin  = (l & 1) ? g1 : g0;
        half_t* gout = (l & 1) ? g0 : g1;
        gemm_wmma<12, 1, false, false><<<dim3(3, N / 128), blk, 0, stream>>>(
            gin, 384, pG + (size_t)l * 12 * 12 * 512, bg + (size_t)l * GWIDTH,
            nullptr, nullptr, nullptr, 0, gout, 384, 12);
        if (l < GLAYERS - 1)
            gather_mean<<<N, GWIDTH, 0, stream>>>(knn, gout);
    }
    // final gfeat lives in g0 (layer 3 wrote g0)

    // h = h0 + gfeat @ Wg_out   (reads g0, writes hA==g1: safe)
    gemm_wmma<6, 0, false, true><<<dim3(4, N / 128), blk, 0, stream>>>(
        g0, 384, pWgo, nullptr, nullptr, nullptr, h0h, WIDTH, hA, WIDTH, 16);

    // FiLM trunk
    half_t* cur = hA;
    half_t* nxt = hB;
    for (int l = 0; l < DEPTH; ++l) {
        const float* ga = gb + l * 512;
        const float* be = gb + l * 512 + 256;
        const float* bi = bmlp + (size_t)l * WIDTH;
        half_t* bw = pW + (size_t)l * 8 * 16 * 512;
        const bool skip = (l == 2) || (l == 5);   // (l+1) in {3,6}
        if (skip)
            gemm_wmma<8, 1, true, true><<<dim3(4, N / 128), blk, 0, stream>>>(
                cur, WIDTH, bw, bi, ga, be, h0h, WIDTH, nxt, WIDTH, 16);
        else
            gemm_wmma<8, 1, true, false><<<dim3(4, N / 128), blk, 0, stream>>>(
                cur, WIDTH, bw, bi, ga, be, nullptr, 0, nxt, WIDTH, 16);
        half_t* t = cur; cur = nxt; nxt = t;
    }

    // head
    out_kernel<<<N / 256, blk, 0, stream>>>(cur, W_out, b_out, out, N);
}
